// Swin3DBlock_Fixed_56109452755124
// MI455X (gfx1250) — matrix operand order repaired and checked by reading, not compiled, not run on hardware
//
#include <hip/hip_runtime.h>
#include <hip/hip_bf16.h>

// ---------------------------------------------------------------------------
// Swin3D block for MI455X (gfx1250, wave32, WMMA 16x16x32 bf16)
//  - all WMMA fragments are single 32-byte-per-lane vector loads (b128 pairs)
//  - weights pre-packed to fragment tiles; activations in natural k order (no swizzle)
//  - GEMM waves carry 4 N-accumulators (4x A reuse, 4 WMMAs per K step)
//  - attention: scores never hit HBM; bias+mask+pad fused into one L2-resident
//    table indexed by (window-category[3b], head, m, n)
// ---------------------------------------------------------------------------
constexpr int kB = 2, kC = 128, kHeads = 4;
constexpr int kD = 16, kH = 56, kW = 56;
constexpr int kWD = 8, kWH = 7, kWW = 7;
constexpr int kSD = 4, kSH = 3, kSW = 3;
constexpr int kND = 2, kNH = 8, kNW = 8;
constexpr int kNWin = kND * kNH * kNW;       // 128 windows / batch
constexpr int kBW = kB * kNWin;              // 256 windows total
constexpr int kN = kWD * kWH * kWW;          // 392 tokens / window
constexpr int kNP = 416;                     // padded to 26*16
constexpr int kHd = kC / kHeads;             // 32 (== WMMA K in one shot)
constexpr int kDHW = kD * kH * kW;           // 50176
constexpr int kTok = kB * kDHW;              // 100352 (multiple of 16)
constexpr int kMwin = kBW * kNP;             // 106496 padded window rows
constexpr int kC4 = 4 * kC;                  // 512
constexpr int kKT = kNP / 32;                // 13 V k-tiles

typedef __attribute__((ext_vector_type(16))) __bf16 v16bf;
typedef __attribute__((ext_vector_type(8)))  float  v8f;

// ---- helpers ---------------------------------------------------------------
__device__ __forceinline__ unsigned short f2bf(float f) {
  unsigned int u = __float_as_uint(f);
  u += 0x7FFFu + ((u >> 16) & 1u);           // round-to-nearest-even
  return (unsigned short)(u >> 16);
}
// identity K mapping (contiguous k order within a 32-block)
__device__ __forceinline__ int aswz(int k) {
  return k;
}
__device__ __forceinline__ v16bf ldv(const unsigned short* p) {
  return *reinterpret_cast<const v16bf*>(p);
}
__device__ __forceinline__ v8f wmma_bf16(v16bf a, v16bf b, v8f c) {
  return __builtin_amdgcn_wmma_f32_16x16x32_bf16(
      false, a, false, b, (short)0, c, false, false);
}
// A fragment: source rows in natural k order; per-lane 32B contiguous.
__device__ __forceinline__ v16bf load_a_pk(const unsigned short* __restrict__ A,
                                           int lda, int lane) {
  int m = lane & 15, half = lane >> 4;
  return ldv(A + (size_t)m * lda + 16 * half);
}
// B fragment from packed tile (lane-major, 16 contiguous bf16 per lane).
__device__ __forceinline__ v16bf load_b_pk(const unsigned short* __restrict__ P,
                                           int lane) {
  return ldv(P + (size_t)lane * 16);
}
// B = K^T from natural K [N,32]: contiguous 16 per lane.
__device__ __forceinline__ v16bf load_b_kt(const unsigned short* __restrict__ Kp,
                                           int lane) {
  int n = lane & 15, half = lane >> 4;
  return ldv(Kp + (size_t)n * kHd + 16 * half);
}

// ---------------------------------------------------------------------------
// K0a: pack f32 weights [OUTF, INF] -> bf16 fragment tiles of B=[K=INF,N=OUTF]
// dst index = ((kt*ntiles + nt)*32 + lane)*16 + e ;  k = kt*32+16*half+e
// ---------------------------------------------------------------------------
__global__ void wconv_pack_kernel(const float* __restrict__ src,
                                  unsigned short* __restrict__ dst,
                                  int OUTF, int INF) {
  int i = blockIdx.x * 256 + threadIdx.x;
  if (i >= OUTF * INF) return;
  int e = i & 15, t = i >> 4;
  int lane = t & 31; t >>= 5;
  int ntiles = OUTF >> 4;
  int nt = t % ntiles, kt = t / ntiles;
  int half = lane >> 4, n = lane & 15;
  int k = kt * 32 + 16 * half + e;
  int o = nt * 16 + n;
  dst[i] = f2bf(src[(size_t)o * INF + k]);
}

// ---------------------------------------------------------------------------
// K0b: fused (rel-pos bias + shift mask + key padding) table
// T[cmb][head][m][n] , cmb = ((nd==1)<<2)|((nh==7)<<1)|(nw==7)
// ---------------------------------------------------------------------------
__global__ void bias_table_kernel(const float* __restrict__ rpb,
                                  float* __restrict__ T) {
  int i = blockIdx.x * 256 + threadIdx.x;
  if (i >= 8 * kHeads * kN * kNP) return;
  int n = i % kNP; int t = i / kNP;
  int m = t % kN; t /= kN;
  int head = t & 3, cmb = t >> 2;
  float val;
  if (n >= kN) {
    val = -1e30f;                            // padded key column
  } else {
    int wdm = m / 49, rm = m - wdm * 49, whm = rm / 7, wwm = rm - whm * 7;
    int wdn = n / 49, rn = n - wdn * 49, whn = rn / 7, wwn = rn - whn * 7;
    int idx = (wdm - wdn + 7) * 169 + (whm - whn + 6) * 13 + (wwm - wwn + 6);
    float bias = rpb[(size_t)idx * kHeads + head];
    int dbit = (cmb >> 2) & 1, hbit = (cmb >> 1) & 1, wbit = cmb & 1;
    int am = dbit ? (wdm < kWD - kSD ? 1 : 2) : 0;
    int an = dbit ? (wdn < kWD - kSD ? 1 : 2) : 0;
    int bm = hbit ? (whm < kWH - kSH ? 1 : 2) : 0;
    int bn = hbit ? (whn < kWH - kSH ? 1 : 2) : 0;
    int cm = wbit ? (wwm < kWW - kSW ? 1 : 2) : 0;
    int cn = wbit ? (wwn < kWW - kSW ? 1 : 2) : 0;
    bool same = (am == an) && (bm == bn) && (cm == cn);
    val = bias + (same ? 0.f : -100.f);
  }
  T[i] = val;
}

// ---------------------------------------------------------------------------
// K1: LN1 (over C) + roll(-S) + window partition -> Xwin bf16 [BW,NP,C] (natural c order)
// ---------------------------------------------------------------------------
__global__ __launch_bounds__(256) void ln1_window_kernel(
    const float* __restrict__ x, const float* __restrict__ gw,
    const float* __restrict__ gb, unsigned short* __restrict__ Xwin) {
  int wave = threadIdx.x >> 5, lane = threadIdx.x & 31;
  int t = blockIdx.x * 8 + wave;
  if (t >= kTok) return;
  int b = t / kDHW, s = t - b * kDHW;
  int dr = s / (kH * kW), s2 = s - dr * kH * kW;
  int hr = s2 / kW, wr = s2 - hr * kW;
  int dsrc = (dr + kSD) & (kD - 1);
  int hsrc = hr + kSH; if (hsrc >= kH) hsrc -= kH;
  int wsrc = wr + kSW; if (wsrc >= kW) wsrc -= kW;
  const float* xp = x + (size_t)b * kC * kDHW + (size_t)dsrc * kH * kW + hsrc * kW + wsrc;
  float v[4], sum = 0.f, sq = 0.f;
#pragma unroll
  for (int i = 0; i < 4; ++i) {
    float f = xp[(size_t)(lane + 32 * i) * kDHW];
    v[i] = f; sum += f; sq += f * f;
  }
#pragma unroll
  for (int o2 = 16; o2; o2 >>= 1) { sum += __shfl_xor(sum, o2, 32); sq += __shfl_xor(sq, o2, 32); }
  float mu = sum * (1.f / kC);
  float rstd = rsqrtf(sq * (1.f / kC) - mu * mu + 1e-5f);
  int nd = dr >> 3, wd = dr & 7;
  int nh = hr / 7, wh = hr - nh * 7;
  int nw = wr / 7, ww = wr - nw * 7;
  int win = ((b * kND + nd) * kNH + nh) * kNW + nw;
  int n = (wd * 7 + wh) * 7 + ww;
  unsigned short* op = Xwin + ((size_t)win * kNP + n) * kC;
#pragma unroll
  for (int i = 0; i < 4; ++i) {
    int c = lane + 32 * i;
    op[aswz(c)] = f2bf((v[i] - mu) * rstd * gw[c] + gb[c]);
  }
}

// ---------------------------------------------------------------------------
// K2: QKV GEMM [kMwin,128]x[128,384], 4 N-tiles per wave
// ---------------------------------------------------------------------------
__global__ __launch_bounds__(32) void gemm_qkv_kernel(
    const unsigned short* __restrict__ A, const unsigned short* __restrict__ Bpk,
    const float* __restrict__ qkvb, unsigned short* __restrict__ Qb,
    unsigned short* __restrict__ Kb, unsigned short* __restrict__ Vb) {
  constexpr int ntiles = 3 * kC / 16;        // 24
  int mt = blockIdx.x, nt0 = blockIdx.y * 4, lane = threadIdx.x;
  const unsigned short* Ap = A + (size_t)mt * 16 * kC;
  v8f acc[4] = {};
#pragma unroll
  for (int kc = 0; kc < 4; ++kc) {
    if (kc + 1 < 4) __builtin_prefetch((const void*)(Ap + (kc + 1) * 32), 0, 0);
    v16bf a = load_a_pk(Ap + kc * 32, kC, lane);
#pragma unroll
    for (int u = 0; u < 4; ++u) {
      v16bf b = load_b_pk(Bpk + ((size_t)(kc * ntiles + nt0 + u) * 32) * 16, lane);
      acc[u] = wmma_bf16(a, b, acc[u]);
    }
  }
  int col = lane & 15, half = lane >> 4;
#pragma unroll
  for (int u = 0; u < 4; ++u) {
    int o = (nt0 + u) * 16 + col;
    float bb = qkvb[o];
#pragma unroll
    for (int r = 0; r < 8; ++r) {
      int row = mt * 16 + r + 8 * half;
      int win = row / kNP, n = row - win * kNP;
      float val = acc[u][r] + bb;
      if (o < kC) {                          // Q : natural rows
        int d = o & 31;
        Qb[(((size_t)win * kHeads + (o >> 5)) * kNP + n) * kHd + aswz(d)] =
            f2bf(val * 0.17677669529663687f);
      } else if (o < 2 * kC) {               // K : natural rows
        int oo = o - kC;
        Kb[(((size_t)win * kHeads + (oo >> 5)) * kNP + n) * kHd + (oo & 31)] = f2bf(val);
      } else {                               // V : fragment-packed tiles
        int oo = o - 2 * kC;
        int d = oo & 31, hh = oo >> 5;
        int kt = n >> 5, kk = n & 31;
        int halfk = kk >> 4, rem = kk & 15;
        int lanev = halfk * 16 + (d & 15);
        int ntv = d >> 4;
        Vb[(((((size_t)win * kHeads + hh) * kKT + kt) * 2 + ntv) * 32 + lanev) * 16 + rem] =
            f2bf(val);
      }
    }
  }
}

// ---------------------------------------------------------------------------
// K3: fused attention per (window, head, 16-row block)
// ---------------------------------------------------------------------------
__global__ __launch_bounds__(32) void attn_kernel(
    const unsigned short* __restrict__ Qb, const unsigned short* __restrict__ Kb,
    const unsigned short* __restrict__ Vb, const float* __restrict__ Tbl,
    unsigned short* __restrict__ O) {
  __shared__ __align__(32) float Ssh[16][kNP];
  __shared__ __align__(32) unsigned short Pb[16][kNP];
  int mt = blockIdx.x, head = blockIdx.y, win = blockIdx.z;
  int lane = threadIdx.x;
  int wloc = win & (kNWin - 1);
  int nd = wloc >> 6, nh = (wloc >> 3) & 7, nw = wloc & 7;
  int cmb = ((nd == 1) << 2) | ((nh == 7) << 1) | (nw == 7);
  const float* Trow = Tbl + ((size_t)(cmb * kHeads + head) * kN) * kNP;
  size_t base = ((size_t)win * kHeads + head) * kNP;
  const unsigned short* Qp = Qb + (base + (size_t)mt * 16) * kHd;
  v16bf aq = load_a_pk(Qp, kHd, lane);
  int n15 = lane & 15, half = lane >> 4;
  // clamp row index for the bias table (out-of-range rows are discarded later)
  int mclamp[8];
#pragma unroll
  for (int r = 0; r < 8; ++r) {
    int mrow = mt * 16 + r + 8 * half;
    mclamp[r] = mrow < kN ? mrow : kN - 1;
  }

  for (int j = 0; j < kNP / 16; ++j) {
    const unsigned short* Kp = Kb + (base + (size_t)j * 16) * kHd;
    v16bf bk = load_b_kt(Kp, lane);
    v8f c = {};
    c = wmma_bf16(aq, bk, c);
    int ncol = j * 16 + n15;
#pragma unroll
    for (int r = 0; r < 8; ++r) {
      int lr = r + 8 * half;
      Ssh[lr][ncol] = c[r] + Trow[(size_t)mclamp[r] * kNP + ncol];
    }
  }
  // single wave: per-wave LDS ordering, no barrier needed
  for (int rr = 0; rr < 16; ++rr) {
    float mx = -3.4e38f;
    for (int ci = lane; ci < kNP; ci += 32) mx = fmaxf(mx, Ssh[rr][ci]);
#pragma unroll
    for (int o2 = 16; o2; o2 >>= 1) mx = fmaxf(mx, __shfl_xor(mx, o2, 32));
    float sm = 0.f;
    for (int ci = lane; ci < kNP; ci += 32) {
      float e = __expf(Ssh[rr][ci] - mx);
      Ssh[rr][ci] = e; sm += e;
    }
#pragma unroll
    for (int o2 = 16; o2; o2 >>= 1) sm += __shfl_xor(sm, o2, 32);
    float inv = 1.f / sm;
    for (int ci = lane; ci < kNP; ci += 32)
      Pb[rr][aswz(ci)] = f2bf(Ssh[rr][ci] * inv);   // store P in natural k order
  }
  // O = P @ V : 13 chunks of 32; head dim 32 -> two 16-wide tiles
  v8f acc0 = {}, acc1 = {};
  const unsigned short* Vp = Vb + (((size_t)win * kHeads + head) * kKT * 2 * 32) * 16;
  for (int kc = 0; kc < kKT; ++kc) {
    v16bf a  = load_a_pk(&Pb[0][0] + kc * 32, kNP, lane);
    v16bf b0 = load_b_pk(Vp + ((size_t)(kc * 2 + 0) * 32) * 16, lane);
    v16bf b1 = load_b_pk(Vp + ((size_t)(kc * 2 + 1) * 32) * 16, lane);
    acc0 = wmma_bf16(a, b0, acc0);
    acc1 = wmma_bf16(a, b1, acc1);
  }
#pragma unroll
  for (int r = 0; r < 8; ++r) {
    int mrow = mt * 16 + r + 8 * half;
    size_t orow = ((size_t)win * kNP + mrow) * kC + head * kHd;
    O[orow + aswz(n15)]      = f2bf(acc0[r]);       // natural k order for proj A
    O[orow + aswz(16 + n15)] = f2bf(acc1[r]);
  }
}

// ---------------------------------------------------------------------------
// K4: proj GEMM [kMwin,128]x[128,128], 4 N-tiles/wave; reverse+roll+residual
// ---------------------------------------------------------------------------
__global__ __launch_bounds__(32) void gemm_proj_kernel(
    const unsigned short* __restrict__ A, const unsigned short* __restrict__ Bpk,
    const float* __restrict__ pb, const float* __restrict__ x,
    float* __restrict__ X1) {
  constexpr int ntiles = kC / 16;            // 8
  int mt = blockIdx.x, nt0 = blockIdx.y * 4, lane = threadIdx.x;
  const unsigned short* Ap = A + (size_t)mt * 16 * kC;
  v8f acc[4] = {};
#pragma unroll
  for (int kc = 0; kc < 4; ++kc) {
    if (kc + 1 < 4) __builtin_prefetch((const void*)(Ap + (kc + 1) * 32), 0, 0);
    v16bf a = load_a_pk(Ap + kc * 32, kC, lane);
#pragma unroll
    for (int u = 0; u < 4; ++u) {
      v16bf b = load_b_pk(Bpk + ((size_t)(kc * ntiles + nt0 + u) * 32) * 16, lane);
      acc[u] = wmma_bf16(a, b, acc[u]);
    }
  }
  int col = lane & 15, half = lane >> 4;
#pragma unroll
  for (int r = 0; r < 8; ++r) {
    int row = mt * 16 + r + 8 * half;
    int win = row / kNP, n = row - win * kNP;
    if (n >= kN) continue;                     // padded token rows dropped
    int b_ = win >> 7, wloc = win & (kNWin - 1);
    int nd = wloc >> 6, nh = (wloc >> 3) & 7, nw = wloc & 7;
    int wd = n / 49, rm = n - wd * 49, wh = rm / 7, ww = rm - wh * 7;
    int dr = nd * 8 + wd, hr = nh * 7 + wh, wr = nw * 7 + ww;
    int dro = (dr + kSD) & (kD - 1);
    int hro = hr + kSH; if (hro >= kH) hro -= kH;
    int wro = wr + kSW; if (wro >= kW) wro -= kW;
    size_t sidx = (size_t)dro * kH * kW + hro * kW + wro;
#pragma unroll
    for (int u = 0; u < 4; ++u) {
      int o = (nt0 + u) * 16 + col;
      size_t idx = ((size_t)b_ * kC + o) * kDHW + sidx;
      X1[idx] = x[idx] + acc[u][r] + pb[o];
    }
  }
}

// ---------------------------------------------------------------------------
// K5a: LN2 over C -> token-major bf16 [kTok,128] (natural c order)
// ---------------------------------------------------------------------------
__global__ __launch_bounds__(256) void ln2_kernel(
    const float* __restrict__ X1, const float* __restrict__ gw,
    const float* __restrict__ gb, unsigned short* __restrict__ Xn) {
  int wave = threadIdx.x >> 5, lane = threadIdx.x & 31;
  int t = blockIdx.x * 8 + wave;
  if (t >= kTok) return;
  int b = t / kDHW, s = t - b * kDHW;
  const float* xp = X1 + (size_t)b * kC * kDHW + s;
  float v[4], sum = 0.f, sq = 0.f;
#pragma unroll
  for (int i = 0; i < 4; ++i) {
    float f = xp[(size_t)(lane + 32 * i) * kDHW];
    v[i] = f; sum += f; sq += f * f;
  }
#pragma unroll
  for (int o2 = 16; o2; o2 >>= 1) { sum += __shfl_xor(sum, o2, 32); sq += __shfl_xor(sq, o2, 32); }
  float mu = sum * (1.f / kC);
  float rstd = rsqrtf(sq * (1.f / kC) - mu * mu + 1e-5f);
  unsigned short* op = Xn + (size_t)t * kC;
#pragma unroll
  for (int i = 0; i < 4; ++i) {
    int c = lane + 32 * i;
    op[aswz(c)] = f2bf((v[i] - mu) * rstd * gw[c] + gb[c]);
  }
}

// ---------------------------------------------------------------------------
// K5b: fc1 GEMM [kTok,128]x[128,512], 4 N-tiles/wave, gelu -> H2 (natural order)
// ---------------------------------------------------------------------------
__global__ __launch_bounds__(32) void gemm_fc1_kernel(
    const unsigned short* __restrict__ A, const unsigned short* __restrict__ Bpk,
    const float* __restrict__ fb, unsigned short* __restrict__ H2) {
  constexpr int ntiles = kC4 / 16;           // 32
  int mt = blockIdx.x, nt0 = blockIdx.y * 4, lane = threadIdx.x;
  const unsigned short* Ap = A + (size_t)mt * 16 * kC;
  v8f acc[4] = {};
#pragma unroll
  for (int kc = 0; kc < 4; ++kc) {
    if (kc + 1 < 4) __builtin_prefetch((const void*)(Ap + (kc + 1) * 32), 0, 0);
    v16bf a = load_a_pk(Ap + kc * 32, kC, lane);
#pragma unroll
    for (int u = 0; u < 4; ++u) {
      v16bf b = load_b_pk(Bpk + ((size_t)(kc * ntiles + nt0 + u) * 32) * 16, lane);
      acc[u] = wmma_bf16(a, b, acc[u]);
    }
  }
  int col = lane & 15, half = lane >> 4;
#pragma unroll
  for (int u = 0; u < 4; ++u) {
    int o = (nt0 + u) * 16 + col;
    float bb = fb[o];
    int oswz = aswz(o);
#pragma unroll
    for (int r = 0; r < 8; ++r) {
      int row = mt * 16 + r + 8 * half;
      float g = acc[u][r] + bb;
      g = 0.5f * g * (1.f + tanhf(0.7978845608028654f * (g + 0.044715f * g * g * g)));
      H2[(size_t)row * kC4 + oswz] = f2bf(g);
    }
  }
}

// ---------------------------------------------------------------------------
// K6: fc2 GEMM [kTok,512]x[512,128], 4 N-tiles/wave, + residual -> d_out f32
// ---------------------------------------------------------------------------
__global__ __launch_bounds__(32) void gemm_fc2_kernel(
    const unsigned short* __restrict__ A, const unsigned short* __restrict__ Bpk,
    const float* __restrict__ fb, const float* __restrict__ X1,
    float* __restrict__ out) {
  constexpr int ntiles = kC / 16;            // 8
  int mt = blockIdx.x, nt0 = blockIdx.y * 4, lane = threadIdx.x;
  const unsigned short* Ap = A + (size_t)mt * 16 * kC4;
  v8f acc[4] = {};
#pragma unroll
  for (int kc = 0; kc < 16; ++kc) {
    if (kc + 1 < 16) __builtin_prefetch((const void*)(Ap + (kc + 1) * 32), 0, 0);
    v16bf a = load_a_pk(Ap + kc * 32, kC4, lane);
#pragma unroll
    for (int u = 0; u < 4; ++u) {
      v16bf b = load_b_pk(Bpk + ((size_t)(kc * ntiles + nt0 + u) * 32) * 16, lane);
      acc[u] = wmma_bf16(a, b, acc[u]);
    }
  }
  int col = lane & 15, half = lane >> 4;
#pragma unroll
  for (int r = 0; r < 8; ++r) {
    int t = mt * 16 + r + 8 * half;
    int b_ = t / kDHW, s = t - b_ * kDHW;
#pragma unroll
    for (int u = 0; u < 4; ++u) {
      int o = (nt0 + u) * 16 + col;
      size_t idx = ((size_t)b_ * kC + o) * kDHW + s;
      out[idx] = X1[idx] + acc[u][r] + fb[o];
    }
  }
}

// ---------------------------------------------------------------------------
extern "C" void kernel_launch(void* const* d_in, const int* in_sizes, int n_in,
                              void* d_out, int out_size, void* d_ws, size_t ws_size,
                              hipStream_t stream) {
  const float* x       = (const float*)d_in[0];
  const float* norm1_w = (const float*)d_in[1];
  const float* norm1_b = (const float*)d_in[2];
  const float* qkv_w   = (const float*)d_in[3];
  const float* qkv_b   = (const float*)d_in[4];
  const float* rpb     = (const float*)d_in[5];
  const float* proj_w  = (const float*)d_in[6];
  const float* proj_b  = (const float*)d_in[7];
  const float* norm2_w = (const float*)d_in[8];
  const float* norm2_b = (const float*)d_in[9];
  const float* fc1_w   = (const float*)d_in[10];
  const float* fc1_b   = (const float*)d_in[11];
  const float* fc2_w   = (const float*)d_in[12];
  const float* fc2_b   = (const float*)d_in[13];
  float* out = (float*)d_out;

  char* ws = (char*)d_ws;
  size_t off = 0;
  auto take = [&](size_t bytes) -> char* {
    char* p = ws + off;
    off += (bytes + 255) & ~size_t(255);
    return p;
  };
  unsigned short* Bqkv  = (unsigned short*)take((size_t)kC * 3 * kC * 2);
  unsigned short* Bproj = (unsigned short*)take((size_t)kC * kC * 2);
  unsigned short* Bfc1  = (unsigned short*)take((size_t)kC * kC4 * 2);
  unsigned short* Bfc2  = (unsigned short*)take((size_t)kC4 * kC * 2);
  float*          Tbl   = (float*)take((size_t)8 * kHeads * kN * kNP * 4);
  unsigned short* Xwin  = (unsigned short*)take((size_t)kMwin * kC * 2);
  unsigned short* Qb    = (unsigned short*)take((size_t)kBW * kHeads * kNP * kHd * 2);
  unsigned short* Kb    = (unsigned short*)take((size_t)kBW * kHeads * kNP * kHd * 2);
  unsigned short* Vb    = (unsigned short*)take((size_t)kBW * kHeads * kNP * kHd * 2);
  float*          X1    = (float*)take((size_t)kB * kC * kDHW * 4);
  unsigned short* H2    = (unsigned short*)take((size_t)kTok * kC4 * 2);
  unsigned short* AttnO = Xwin;   // alias: Xwin dead after QKV GEMM
  unsigned short* Xn    = Qb;     // alias: Q/K/V dead after attention

  // K0: weight packing + fused bias/mask table
  wconv_pack_kernel<<<(3 * kC * kC + 255) / 256, 256, 0, stream>>>(qkv_w, Bqkv, 3 * kC, kC);
  wconv_pack_kernel<<<(kC * kC + 255) / 256, 256, 0, stream>>>(proj_w, Bproj, kC, kC);
  wconv_pack_kernel<<<(kC4 * kC + 255) / 256, 256, 0, stream>>>(fc1_w, Bfc1, kC4, kC);
  wconv_pack_kernel<<<(kC * kC4 + 255) / 256, 256, 0, stream>>>(fc2_w, Bfc2, kC, kC4);
  bias_table_kernel<<<(8 * kHeads * kN * kNP + 255) / 256, 256, 0, stream>>>(rpb, Tbl);

  // K1: LN1 + shift + window partition
  ln1_window_kernel<<<kTok / 8, 256, 0, stream>>>(x, norm1_w, norm1_b, Xwin);

  // K2: QKV projection
  gemm_qkv_kernel<<<dim3(kMwin / 16, 3 * kC / 64), 32, 0, stream>>>(
      Xwin, Bqkv, qkv_b, Qb, Kb, Vb);

  // K3: fused windowed attention
  attn_kernel<<<dim3(kNP / 16, kHeads, kBW), 32, 0, stream>>>(Qb, Kb, Vb, Tbl, AttnO);

  // K4: output projection + window reverse + residual
  gemm_proj_kernel<<<dim3(kMwin / 16, kC / 64), 32, 0, stream>>>(
      AttnO, Bproj, proj_b, x, X1);

  // K5: LN2 + FFN fc1 (gelu)
  ln2_kernel<<<kTok / 8, 256, 0, stream>>>(X1, norm2_w, norm2_b, Xn);
  gemm_fc1_kernel<<<dim3(kTok / 16, kC4 / 64), 32, 0, stream>>>(Xn, Bfc1, fc1_b, H2);

  // K6: FFN fc2 + residual -> output
  gemm_fc2_kernel<<<dim3(kTok / 16, kC / 64), 32, 0, stream>>>(H2, Bfc2, fc2_b, X1, out);
}
